// Attention_8418135900661
// MI455X (gfx1250) — matrix operand order repaired and checked by reading, not compiled, not run on hardware
//
#include <hip/hip_runtime.h>
#include <hip/hip_bf16.h>

// Problem constants (reference: T=4096, D=1024, H=16, HS=64)
#define T_DIM  4096
#define D_DIM  1024
#define H_DIM  16
#define HS_DIM 64

typedef __attribute__((ext_vector_type(16))) __bf16 v16bf;
typedef __attribute__((ext_vector_type(8)))  __bf16 v8bf;
typedef __attribute__((ext_vector_type(8)))  float  v8f;

// Pointer types for the async-LDS builtin: param is vector_size(16) int,
// pointee in AS1 (global, printed as __device__) / AS3 (LDS).
typedef int vs4i __attribute__((vector_size(16)));
typedef __attribute__((address_space(1))) vs4i* g_v4i_ptr;
typedef __attribute__((address_space(3))) vs4i* l_v4i_ptr;

union ABTile { v16bf v; v8bf h[2]; };

__device__ __forceinline__ v8f wmma_bf16(v16bf a, v16bf b, v8f c) {
  // (neg_a, A, neg_b, B, c_mod, C, reuse_a, reuse_b)
  return __builtin_amdgcn_wmma_f32_16x16x32_bf16(false, a, false, b, (short)0, c,
                                                 false, false);
}

// ---------------------------------------------------------------------------
// Async global -> LDS copy of 16 bytes per lane (GLOBAL_LOAD_ASYNC_TO_LDS_B128,
// tracked by ASYNCcnt). Builtin confirmed present with prototype
// (global v4i*, local v4i*, imm int, imm int).
// ---------------------------------------------------------------------------
__device__ __forceinline__ void async_copy_b128(const __bf16* g, __bf16* l) {
#if __has_builtin(__builtin_amdgcn_global_load_async_to_lds_b128)
  __builtin_amdgcn_global_load_async_to_lds_b128(
      (g_v4i_ptr)(void*)g, (l_v4i_ptr)(void*)l, 0, 0);
#else
  unsigned int loff =
      (unsigned int)(uintptr_t)(__attribute__((address_space(3))) void*)(void*)l;
  unsigned long long ga = (unsigned long long)(uintptr_t)g;
  asm volatile("global_load_async_to_lds_b128 %0, %1, off"
               :: "v"(loff), "v"(ga) : "memory");
#endif
}

__device__ __forceinline__ void wait_asynccnt0() {
#if __has_builtin(__builtin_amdgcn_s_wait_asynccnt)
  __builtin_amdgcn_s_wait_asynccnt(0);
#else
  asm volatile("s_wait_asynccnt 0" ::: "memory");
#endif
}

// ---------------------------------------------------------------------------
// Kernel 0a: fp32 -> bf16 (x, row-major [T][D])
// ---------------------------------------------------------------------------
__global__ __launch_bounds__(256) void cvt_x_kernel(const float* __restrict__ src,
                                                    __bf16* __restrict__ dst, int n) {
  int i = blockIdx.x * 256 + threadIdx.x;
  int stride = gridDim.x * 256;
  for (; i < n; i += stride) dst[i] = (__bf16)src[i];
}

// ---------------------------------------------------------------------------
// Kernel 0b: W [H][D][HS] fp32 -> Wt [H][HS][D] bf16 (transpose + convert)
// so WMMA B-operand tiles become two contiguous 16B runs along D per lane.
// ---------------------------------------------------------------------------
__global__ __launch_bounds__(256) void cvt_w_kernel(const float* __restrict__ src,
                                                    __bf16* __restrict__ dst) {
  const int n = H_DIM * D_DIM * HS_DIM;
  int i = blockIdx.x * 256 + threadIdx.x;
  int stride = gridDim.x * 256;
  for (; i < n; i += stride) {
    int s = i % HS_DIM;
    int d = (i / HS_DIM) % D_DIM;
    int h = i / (HS_DIM * D_DIM);
    dst[((size_t)h * HS_DIM + s) * D_DIM + d] = (__bf16)src[i];
  }
}

// ---------------------------------------------------------------------------
// Kernel 1: projections. One wave (32 lanes) per (16-row tile, head, z).
//   z=0: Kb[h][t][s]   z=1: Qb[h][t][s]   z=2: Vt[h][s][t]  (all bf16)
// Out[t][s] = sum_d xb[t][d] * Wt[h][s][d], accumulated with
// v_wmma_f32_16x16x32_bf16 over D in k-steps of 32.
// ---------------------------------------------------------------------------
__global__ __launch_bounds__(32) void proj_kernel(
    const __bf16* __restrict__ xb,
    const __bf16* __restrict__ Wkt, const __bf16* __restrict__ Wqt,
    const __bf16* __restrict__ Wvt,
    __bf16* __restrict__ Kb, __bf16* __restrict__ Qb, __bf16* __restrict__ Vt) {
  const int tile  = blockIdx.x;       // 0..T/16-1
  const int h     = blockIdx.y;       // 0..H-1
  const int z     = blockIdx.z;       // 0=K, 1=Q, 2=V
  const int lane  = threadIdx.x;      // 0..31
  const int laneN = lane & 15;
  const int hi    = lane >> 4;        // half-wave select
  const int i0    = tile * 16;

  const __bf16* Wt = (z == 0) ? Wkt : (z == 1) ? Wqt : Wvt;
  const __bf16* xrow  = xb + (size_t)(i0 + laneN) * D_DIM;
  const __bf16* wbase = Wt + (size_t)h * HS_DIM * D_DIM;

  v8f acc[4];
#pragma unroll
  for (int nt = 0; nt < 4; ++nt)
#pragma unroll
    for (int e = 0; e < 8; ++e) acc[nt][e] = 0.0f;

  for (int dk = 0; dk < D_DIM; dk += 32) {
    // A = x tile 16x32 (lane m holds row m; K pairs per ISA layout)
    ABTile a;
    a.h[0] = *(const v8bf*)(xrow + dk + hi * 8);
    a.h[1] = *(const v8bf*)(xrow + dk + 16 + hi * 8);
#pragma unroll
    for (int nt = 0; nt < 4; ++nt) {
      // B[k][n] = W[h][dk+k][nt*16+n] = Wt[h][nt*16+n][dk+k]: K pairs per ISA layout
      ABTile b;
      b.h[0] = *(const v8bf*)(wbase + (size_t)(nt * 16 + laneN) * D_DIM + dk + hi * 8);
      b.h[1] = *(const v8bf*)(wbase + (size_t)(nt * 16 + laneN) * D_DIM + dk + 16 + hi * 8);
      acc[nt] = wmma_bf16(a.v, b.v, acc[nt]);
    }
  }

  // C layout: VGPR r -> row (i0 + r + 8*hi), col = nt*16 + laneN
  if (z == 2) {
#pragma unroll
    for (int nt = 0; nt < 4; ++nt)
#pragma unroll
      for (int r = 0; r < 8; ++r) {
        int row = i0 + r + hi * 8;
        int s   = nt * 16 + laneN;
        Vt[((size_t)h * HS_DIM + s) * T_DIM + row] = (__bf16)acc[nt][r];
      }
  } else {
    __bf16* Out = (z == 0) ? Kb : Qb;
#pragma unroll
    for (int nt = 0; nt < 4; ++nt)
#pragma unroll
      for (int r = 0; r < 8; ++r) {
        int row = i0 + r + hi * 8;
        Out[((size_t)h * T_DIM + row) * HS_DIM + nt * 16 + laneN] = (__bf16)acc[nt][r];
      }
  }
}

// ---------------------------------------------------------------------------
// Kernel 2: causal flash attention (roles: "query" rows come from K).
// Block = 128 threads (4 waves); wave w owns rows [blk*64 + w*16, +16).
// The j-loop is UNIFORM across the block (trip = 2*blk + 2): each iteration
// cooperatively stages the Q-block (32x64) and V-block (64x32) into LDS via
// GLOBAL_LOAD_ASYNC_TO_LDS (ASYNCcnt), then every wave feeds its WMMAs from
// LDS. Waves past their causal diagonal predicate off the compute.
// ---------------------------------------------------------------------------
__global__ __launch_bounds__(128) void attn_kernel(
    const __bf16* __restrict__ Kb, const __bf16* __restrict__ Qb,
    const __bf16* __restrict__ Vt, float* __restrict__ out) {
  __shared__ __bf16 ldsQ[32][HS_DIM];  // Q rows jb..jb+31        (4 KB)
  __shared__ __bf16 ldsV[HS_DIM][32];  // Vt rows s, cols jb..+31 (4 KB)
  __shared__ __bf16 plds[4][16][32];   // per-wave P relayout slab (4 KB)

  const int h     = blockIdx.y;
  const int tid   = threadIdx.x;
  const int lane  = tid & 31;
  const int wave  = tid >> 5;
  const int laneN = lane & 15;
  const int hi    = lane >> 4;
  const int i0    = blockIdx.x * 64 + wave * 16;

  // A-operand (Ke rows), 2 k-steps of 32 over HS=64, loaded once.
  const __bf16* krow = Kb + ((size_t)h * T_DIM + i0 + laneN) * HS_DIM;
  ABTile aK[2];
#pragma unroll
  for (int ks = 0; ks < 2; ++ks) {
    aK[ks].h[0] = *(const v8bf*)(krow + 32 * ks + hi * 8);
    aK[ks].h[1] = *(const v8bf*)(krow + 32 * ks + 16 + hi * 8);
  }

  float m[8], l[8];
  v8f o[4];
#pragma unroll
  for (int r = 0; r < 8; ++r) { m[r] = -__builtin_inff(); l[r] = 0.0f; }
#pragma unroll
  for (int nt = 0; nt < 4; ++nt)
#pragma unroll
    for (int e = 0; e < 8; ++e) o[nt][e] = 0.0f;

  const float scale = 0.125f;  // 1/sqrt(HS)
  const __bf16* qhead = Qb + (size_t)h * T_DIM * HS_DIM;
  const __bf16* vhead = Vt + (size_t)h * HS_DIM * T_DIM;
  const int jend = i0 + 16;                    // per-wave causal bound
  const int jblocks = blockIdx.x * 2 + 2;      // uniform across the block

  // Cooperative staging coordinates for this thread (32 B each):
  const int qrow = tid >> 2, qchunk = tid & 3;  // 32 rows x 4 chunks of 16
  const int vrow = tid >> 1, vhalf  = tid & 1;  // 64 rows x 2 chunks of 16

  for (int jbi = 0; jbi < jblocks; ++jbi) {
    const int jb = jbi * 32;

    // ---- cooperative async stage of Q/V j-block into LDS ----
    {
      const __bf16* gq = qhead + (size_t)(jb + qrow) * HS_DIM + qchunk * 16;
      const __bf16* gv = vhead + (size_t)vrow * T_DIM + jb + vhalf * 16;
      __bf16* lq = &ldsQ[qrow][qchunk * 16];
      __bf16* lv = &ldsV[vrow][vhalf * 16];
      async_copy_b128(gq,     lq);
      async_copy_b128(gq + 8, lq + 8);
      async_copy_b128(gv,     lv);
      async_copy_b128(gv + 8, lv + 8);
    }

    // Warm L2/WGP$ for the next j-block (global_prefetch_b8).
    if (jbi + 1 < jblocks) {
      __builtin_prefetch(qhead + (size_t)(jb + 32 + qrow) * HS_DIM + qchunk * 16, 0, 3);
      __builtin_prefetch(vhead + (size_t)vrow * T_DIM + jb + 32 + vhalf * 16, 0, 3);
    }

    wait_asynccnt0();   // this wave's async transfers landed in LDS
    __syncthreads();    // all waves' transfers visible block-wide

    if (jb < jend) {
      // ---- S = Ke x Qe^T : two 16-col tiles, two k-steps each ----
      v8f s0, s1;
#pragma unroll
      for (int e = 0; e < 8; ++e) { s0[e] = 0.0f; s1[e] = 0.0f; }
#pragma unroll
      for (int ks = 0; ks < 2; ++ks) {
        ABTile b0, b1;
        b0.h[0] = *(const v8bf*)&ldsQ[laneN][32 * ks + hi * 8];
        b0.h[1] = *(const v8bf*)&ldsQ[laneN][32 * ks + 16 + hi * 8];
        b1.h[0] = *(const v8bf*)&ldsQ[16 + laneN][32 * ks + hi * 8];
        b1.h[1] = *(const v8bf*)&ldsQ[16 + laneN][32 * ks + 16 + hi * 8];
        s0 = wmma_bf16(aK[ks].v, b0.v, s0);
        s1 = wmma_bf16(aK[ks].v, b1.v, s1);
      }

      // ---- scale + causal mask + online softmax ----
      float alpha[8];
#pragma unroll
      for (int r = 0; r < 8; ++r) {
        const int row = i0 + r + hi * 8;
        float v0 = s0[r] * scale;
        float v1 = s1[r] * scale;
        if (jb + laneN > row)      v0 = -__builtin_inff();
        if (jb + 16 + laneN > row) v1 = -__builtin_inff();

        float mx = fmaxf(v0, v1);
        mx = fmaxf(mx, __shfl_xor(mx, 1, 32));
        mx = fmaxf(mx, __shfl_xor(mx, 2, 32));
        mx = fmaxf(mx, __shfl_xor(mx, 4, 32));
        mx = fmaxf(mx, __shfl_xor(mx, 8, 32));
        const float mnew = fmaxf(m[r], mx);

        const float a  = __expf(m[r] - mnew);   // 0 when m[r] was -inf
        const float p0 = __expf(v0 - mnew);
        const float p1 = __expf(v1 - mnew);
        float sum = p0 + p1;
        sum += __shfl_xor(sum, 1, 32);
        sum += __shfl_xor(sum, 2, 32);
        sum += __shfl_xor(sum, 4, 32);
        sum += __shfl_xor(sum, 8, 32);

        l[r] = l[r] * a + sum;
        m[r] = mnew;
        alpha[r] = a;

        // P (C-layout element) -> LDS at its (row, col) position
        plds[wave][r + hi * 8][laneN]      = (__bf16)p0;
        plds[wave][r + hi * 8][16 + laneN] = (__bf16)p1;
      }

      // DS ops from the same wave are in-order; fence the compiler + counter.
      asm volatile("s_wait_dscnt 0" ::: "memory");

      // Reload P in A-operand layout: lane m = row m, K-pairs per ISA layout.
      ABTile pA;
      pA.h[0] = *(const v8bf*)&plds[wave][laneN][hi * 8];
      pA.h[1] = *(const v8bf*)&plds[wave][laneN][16 + hi * 8];

      // ---- O = O*alpha + P x V ----
#pragma unroll
      for (int nt = 0; nt < 4; ++nt) {
#pragma unroll
        for (int r = 0; r < 8; ++r) o[nt][r] *= alpha[r];
        // B[k][n] = V[jb+k][nt*16+n] = ldsV[nt*16+n][k]: K pairs per ISA layout
        ABTile bV;
        bV.h[0] = *(const v8bf*)&ldsV[nt * 16 + laneN][hi * 8];
        bV.h[1] = *(const v8bf*)&ldsV[nt * 16 + laneN][16 + hi * 8];
        o[nt] = wmma_bf16(pA.v, bV.v, o[nt]);
      }
    }
    __syncthreads();  // protect ldsQ/ldsV before next stage
  }

  // ---- normalize and store: out[t][h*HS + s], fp32 ----
#pragma unroll
  for (int r = 0; r < 8; ++r) {
    const float inv = 1.0f / l[r];
    const int row = i0 + r + hi * 8;
#pragma unroll
    for (int nt = 0; nt < 4; ++nt) {
      out[(size_t)row * (H_DIM * HS_DIM) + h * HS_DIM + nt * 16 + laneN] =
          o[nt][r] * inv;
    }
  }
}

// ---------------------------------------------------------------------------
// Launcher
// ---------------------------------------------------------------------------
extern "C" void kernel_launch(void* const* d_in, const int* in_sizes, int n_in,
                              void* d_out, int out_size, void* d_ws, size_t ws_size,
                              hipStream_t stream) {
  const float* x  = (const float*)d_in[0];
  const float* Wk = (const float*)d_in[1];
  const float* Wq = (const float*)d_in[2];
  const float* Wv = (const float*)d_in[3];
  float* out = (float*)d_out;

  // Workspace carve-up (bf16 everywhere): ~38 MB total
  char* ws = (char*)d_ws;
  __bf16* xb  = (__bf16*)ws; ws += (size_t)T_DIM * D_DIM * sizeof(__bf16);
  __bf16* Wkt = (__bf16*)ws; ws += (size_t)H_DIM * HS_DIM * D_DIM * sizeof(__bf16);
  __bf16* Wqt = (__bf16*)ws; ws += (size_t)H_DIM * HS_DIM * D_DIM * sizeof(__bf16);
  __bf16* Wvt = (__bf16*)ws; ws += (size_t)H_DIM * HS_DIM * D_DIM * sizeof(__bf16);
  __bf16* Kb  = (__bf16*)ws; ws += (size_t)H_DIM * T_DIM * HS_DIM * sizeof(__bf16);
  __bf16* Qb  = (__bf16*)ws; ws += (size_t)H_DIM * T_DIM * HS_DIM * sizeof(__bf16);
  __bf16* Vt  = (__bf16*)ws; ws += (size_t)H_DIM * HS_DIM * T_DIM * sizeof(__bf16);

  const int nx = T_DIM * D_DIM;
  cvt_x_kernel<<<dim3((nx + 255) / 256), dim3(256), 0, stream>>>(x, xb, nx);

  const int nw = H_DIM * D_DIM * HS_DIM;
  cvt_w_kernel<<<dim3((nw + 255) / 256), dim3(256), 0, stream>>>(Wk, Wkt);
  cvt_w_kernel<<<dim3((nw + 255) / 256), dim3(256), 0, stream>>>(Wq, Wqt);
  cvt_w_kernel<<<dim3((nw + 255) / 256), dim3(256), 0, stream>>>(Wv, Wvt);

  proj_kernel<<<dim3(T_DIM / 16, H_DIM, 3), dim3(32), 0, stream>>>(
      xb, Wkt, Wqt, Wvt, Kb, Qb, Vt);

  attn_kernel<<<dim3(T_DIM / 64, H_DIM), dim3(128), 0, stream>>>(Kb, Qb, Vt, out);
}
